// SchNetFilter_45037027066140
// MI455X (gfx1250) — matrix operand order repaired and checked by reading, not compiled, not run on hardware
//
#include <hip/hip_runtime.h>

typedef __attribute__((ext_vector_type(16))) __bf16 v16bf;
typedef __attribute__((ext_vector_type(8)))  float  v8f;

#define N_IN 128
#define N_F  128
#define TILE_M 128
#define ROWSTRIDE 136   // bf16 units: 128 data + 8 pad (16B) -> 272B row, bank-conflict-free
#define LN2F 0.69314718055994531f

__device__ __forceinline__ unsigned short bf16_bits(float f) {
  __bf16 b = (__bf16)f;
  return __builtin_bit_cast(unsigned short, b);
}

__device__ __forceinline__ uint2 pack4_bf16(float4 f) {
  uint2 r;
  r.x = (unsigned)bf16_bits(f.x) | ((unsigned)bf16_bits(f.y) << 16);
  r.y = (unsigned)bf16_bits(f.z) | ((unsigned)bf16_bits(f.w) << 16);
  return r;
}

union FragU { struct { uint4 lo; uint4 hi; } u; v16bf v; };

// A (16x32 bf16): lane half 0 -> V0-3 = K+0..7, V4-7 = K+8..15 ; half 1 offset +16 K
__device__ __forceinline__ v16bf load_fragA(const unsigned short* p) {
  FragU f;
  f.u.lo = *(const uint4*)(p);       // 8 bf16: K base+0..7
  f.u.hi = *(const uint4*)(p + 8);   // 8 bf16: K base+8..15
  return f.v;
}

// B (32x16 bf16): lane n holds column n; each half holds 16 contiguous K values
__device__ __forceinline__ v16bf load_fragB(const unsigned short* p) {
  FragU f;
  f.u.lo = *(const uint4*)(p);
  f.u.hi = *(const uint4*)(p + 8);
  return f.v;
}

__device__ __forceinline__ float sspf(float x) {
  // shifted softplus: softplus(x) - ln2, numerically stable form
  float t = __expf(-fabsf(x));
  return fmaxf(x, 0.0f) + __logf(1.0f + t) - LN2F;
}

__global__ __launch_bounds__(256) void schnet_prep_weights(
    const float* __restrict__ W1, const float* __restrict__ W2,
    unsigned short* __restrict__ w1t, unsigned short* __restrict__ w2t)
{
  int idx = blockIdx.x * 256 + threadIdx.x;
  if (idx >= 2 * N_IN * N_F) return;
  const float* W = (idx < N_IN * N_F) ? W1 : W2;
  unsigned short* o = (idx < N_IN * N_F) ? w1t : w2t;
  int e = idx & (N_IN * N_F - 1);
  int k = e >> 7, n = e & 127;          // W[k][n] row-major
  o[n * ROWSTRIDE + k] = bf16_bits(W[e]); // store W^T[n][k], padded
}

__global__ __launch_bounds__(256) void schnet_zero(float* __restrict__ out, int n) {
  int i = blockIdx.x * 256 + threadIdx.x;
  if (i < n) out[i] = 0.0f;
}

__global__ __launch_bounds__(256) void schnet_main(
    const float* __restrict__ dijk,
    const float* __restrict__ b1, const float* __restrict__ b2,
    const int* __restrict__ seg,
    const unsigned short* __restrict__ w1t, const unsigned short* __restrict__ w2t,
    float* __restrict__ out, int E)
{
  __shared__ __attribute__((aligned(16))) unsigned short sA [TILE_M * ROWSTRIDE];
  __shared__ __attribute__((aligned(16))) unsigned short sW1[N_F * ROWSTRIDE];
  __shared__ __attribute__((aligned(16))) unsigned short sW2[N_F * ROWSTRIDE];

  const int tid   = threadIdx.x;
  const int wv    = tid >> 5;
  const int ln    = tid & 31;
  const int lhalf = ln >> 4;
  const int l15   = ln & 15;
  const int rowM  = wv * 16;
  const int blockRow0 = blockIdx.x * TILE_M;

  // ---- stage weights (L2-resident bf16) into LDS ----
  {
    const uint4* s1 = (const uint4*)w1t;
    const uint4* s2 = (const uint4*)w2t;
    uint4* d1 = (uint4*)sW1;
    uint4* d2 = (uint4*)sW2;
    const int n4 = (N_F * ROWSTRIDE) / 8; // 2176 uint4 per matrix
    for (int u = tid; u < n4; u += 256) { d1[u] = s1[u]; d2[u] = s2[u]; }
  }
  // ---- stage A tile: 128 rows x 128 f32 -> bf16 in LDS ----
  {
    const float4* din = (const float4*)dijk;
    for (int u = tid; u < TILE_M * 32; u += 256) {
      int row = u >> 5, c4 = u & 31;
      int g = blockRow0 + row;
      float4 f = make_float4(0.0f, 0.0f, 0.0f, 0.0f);
      if (g < E) f = din[g * 32 + c4];
      *(uint2*)((char*)sA + row * (ROWSTRIDE * 2) + c4 * 8) = pack4_bf16(f);
    }
  }
  __syncthreads();

  v8f acc[8];
  const v8f vzero = {0.f, 0.f, 0.f, 0.f, 0.f, 0.f, 0.f, 0.f};

  // ================= layer 1: h = ssp(A @ W1 + b1) =================
  #pragma unroll
  for (int nt = 0; nt < 8; ++nt) acc[nt] = vzero;
  #pragma unroll
  for (int kc = 0; kc < 4; ++kc) {
    const unsigned short* ap = sA + (rowM + l15) * ROWSTRIDE + kc * 32 + lhalf * 16;
    v16bf a = load_fragA(ap);
    #pragma unroll
    for (int nt = 0; nt < 8; ++nt) {
      const unsigned short* bp = sW1 + (nt * 16 + l15) * ROWSTRIDE + kc * 32 + lhalf * 16;
      v16bf b = load_fragB(bp);
      acc[nt] = __builtin_amdgcn_wmma_f32_16x16x32_bf16(
          false, a, false, b, (short)0, acc[nt], false, false);
    }
  }
  // bias + ssp, write h back (bf16) into this wave's own 16 rows of sA.
  // Per-wave LDS ops are in-order and row ranges are wave-disjoint: no barrier needed.
  #pragma unroll
  for (int nt = 0; nt < 8; ++nt) {
    float bias = b1[nt * 16 + l15];
    #pragma unroll
    for (int r = 0; r < 8; ++r) {
      float h = sspf(acc[nt][r] + bias);
      int row = rowM + lhalf * 8 + r;
      sA[row * ROWSTRIDE + nt * 16 + l15] = bf16_bits(h);
    }
  }

  // ================= layer 2: w = ssp(h @ W2 + b2) =================
  #pragma unroll
  for (int nt = 0; nt < 8; ++nt) acc[nt] = vzero;
  #pragma unroll
  for (int kc = 0; kc < 4; ++kc) {
    const unsigned short* ap = sA + (rowM + l15) * ROWSTRIDE + kc * 32 + lhalf * 16;
    v16bf a = load_fragA(ap);
    #pragma unroll
    for (int nt = 0; nt < 8; ++nt) {
      const unsigned short* bp = sW2 + (nt * 16 + l15) * ROWSTRIDE + kc * 32 + lhalf * 16;
      v16bf b = load_fragB(bp);
      acc[nt] = __builtin_amdgcn_wmma_f32_16x16x32_bf16(
          false, a, false, b, (short)0, acc[nt], false, false);
    }
  }

  // ============ bias + ssp + sorted-run segment-sum with atomics ============
  float bias2[8];
  #pragma unroll
  for (int nt = 0; nt < 8; ++nt) bias2[nt] = b2[nt * 16 + l15];

  float run[8];
  #pragma unroll
  for (int nt = 0; nt < 8; ++nt) run[nt] = 0.0f;
  int segPrev = -1;

  // lane walks rows rowBase..rowBase+7 (C-matrix VGPR index r == row offset)
  #pragma unroll
  for (int r = 0; r < 8; ++r) {
    int g = blockRow0 + rowM + lhalf * 8 + r;
    int s = (g < E) ? seg[g] : -1;
    if (s != segPrev) {
      if (segPrev >= 0) {
        #pragma unroll
        for (int nt = 0; nt < 8; ++nt)
          atomicAdd(out + segPrev * N_F + nt * 16 + l15, run[nt]);
      }
      #pragma unroll
      for (int nt = 0; nt < 8; ++nt) run[nt] = 0.0f;
      segPrev = s;
    }
    if (s >= 0) {
      #pragma unroll
      for (int nt = 0; nt < 8; ++nt)
        run[nt] += sspf(acc[nt][r] + bias2[nt]);
    }
  }
  if (segPrev >= 0) {
    #pragma unroll
    for (int nt = 0; nt < 8; ++nt)
      atomicAdd(out + segPrev * N_F + nt * 16 + l15, run[nt]);
  }
}

extern "C" void kernel_launch(void* const* d_in, const int* in_sizes, int n_in,
                              void* d_out, int out_size, void* d_ws, size_t ws_size,
                              hipStream_t stream) {
  const float* dijk = (const float*)d_in[0];
  const float* W1   = (const float*)d_in[1];
  const float* b1   = (const float*)d_in[2];
  const float* W2   = (const float*)d_in[3];
  const float* b2   = (const float*)d_in[4];
  const int*   seg  = (const int*)d_in[5];
  float* out = (float*)d_out;

  int E = in_sizes[0] / N_IN;
  unsigned short* w1t = (unsigned short*)d_ws;
  unsigned short* w2t = w1t + N_F * ROWSTRIDE;

  schnet_prep_weights<<<(2 * N_IN * N_F + 255) / 256, 256, 0, stream>>>(W1, W2, w1t, w2t);
  schnet_zero<<<(out_size + 255) / 256, 256, 0, stream>>>(out, out_size);

  int nblk = (E + TILE_M - 1) / TILE_M;
  schnet_main<<<nblk, 256, 0, stream>>>(dijk, b1, b2, seg, w1t, w2t, out, E);
}
